// Transformer_42494406426931
// MI455X (gfx1250) — matrix operand order repaired and checked by reading, not compiled, not run on hardware
//
#include <hip/hip_runtime.h>
#include <cstdint>
#include <cstddef>

#define NH     4
#define ATT    16
#define COL    64
#define SK     200
#define SK_PAD 208
#define NTILES 13   // 208 / 16

typedef __attribute__((ext_vector_type(16))) __bf16 v16bf;
typedef __attribute__((ext_vector_type(8)))  float  v8f;

// ---------------- LDS layout (bytes) ----------------
// Weight fragments live in LDS in *fragment-native* layout:
//   frag f = kc*4 + nt  (kc: K chunk of 32, nt: N tile of 16)
//   halfword index = f*512 + lane*16 + j   -> per-lane 32B contiguous block
#define OFF_ACT 0
#define OFF_KH  (OFF_ACT + SK_PAD * COL * 4)          // act f32:  53248
#define OFF_VH  (OFF_KH  + SK_PAD * COL * 2)          // kh bf16:  26624
#define OFF_WK  (OFF_VH  + SK_PAD * COL * 2)          // vh bf16:  26624
#define OFF_WV  (OFF_WK  + 8 * 512 * 2)               // K frags:   8192
#define OFF_Q   (OFF_WV  + 8 * 512 * 2)               // V frags:   8192
#define OFF_SC  (OFF_Q   + COL * 4)                   // q  f32:     256
#define OFF_RED (OFF_SC  + NH * SK_PAD * 4)           // scores:    3328
#define LDS_BYTES (OFF_RED + 256 * 4)                 // +red 1024 => 127488 B

__global__ void attn_fused(const float* __restrict__ qin,
                           const float* __restrict__ act,
                           const float* __restrict__ Qw,
                           const float* __restrict__ Kw,
                           const float* __restrict__ Vw,
                           float* __restrict__ out)
{
  extern __shared__ char smem[];
  float*  sAct = (float*)(smem + OFF_ACT);
  __bf16* sKH  = (__bf16*)(smem + OFF_KH);
  __bf16* sVH  = (__bf16*)(smem + OFF_VH);
  __bf16* sWK  = (__bf16*)(smem + OFF_WK);
  __bf16* sWV  = (__bf16*)(smem + OFF_WV);
  float*  sQ   = (float*)(smem + OFF_Q);
  float*  sSc  = (float*)(smem + OFF_SC);
  float*  sRed = (float*)(smem + OFF_RED);

  const int b    = blockIdx.x;
  const int t    = threadIdx.x;
  const int lane = t & 31;
  const int wave = t >> 5;

  // ---- 1) async stage act[b] (200x64 f32, clamp-padded to 208 rows) into LDS ----
  {
    const char* gbase = (const char*)(act + (size_t)b * SK * COL);
    #pragma unroll
    for (int pass = 0; pass < NTILES; ++pass) {
      int ldsRow = pass * 16 + (t >> 4);
      int gRow   = ldsRow < SK ? ldsRow : SK - 1;   // clamped rows are never read
      int colB   = (t & 15) * 16;                   // 16B per lane, 256B per row
      unsigned ldsAddr = (unsigned)(size_t)((char*)sAct + ldsRow * 256 + colB);
      unsigned long long gAddr = (unsigned long long)(gbase + (size_t)gRow * 256 + colB);
      asm volatile("global_load_async_to_lds_b128 %0, %1, off"
                   :: "v"(ldsAddr), "v"(gAddr) : "memory");
    }
  }

  // ---- 2) cooperative fill of K/V weight fragments (bf16) into LDS ----
  // B layout (32x16, 16-bit): lane = N column (mod 16), 16 halves = K rows (table order, +8 for lanes 16..31).
  for (int idx = t; idx < 4096; idx += 256) {
    const int f  = idx >> 9;        // fragment 0..7 (kc*4 + nt)
    const int r  = idx & 511;
    const int ln = r >> 4;          // lane 0..31
    const int j  = r & 15;          // half 0..15
    const int kc = f >> 2;
    const int nt = f & 3;
    const int src = (kc * 32 + ((j < 8) ? 8 * (ln >> 4) + j : 16 + 8 * (ln >> 4) + (j - 8))) * COL + nt * 16 + (ln & 15);
    sWK[idx] = (__bf16)Kw[src];
    sWV[idx] = (__bf16)Vw[src];
  }

  // ---- 3) q projection (tiny GEMV, f32) ----
  if (t < COL) {
    const float* qp = qin + (size_t)b * COL;
    float acc = 0.f;
    #pragma unroll 8
    for (int k = 0; k < COL; ++k) acc += qp[k] * Qw[k * COL + t];
    sQ[t] = acc;
  }

  // ---- wait for async DMA, then make staged data visible to all waves ----
#if __has_builtin(__builtin_amdgcn_s_wait_asynccnt)
  __builtin_amdgcn_s_wait_asynccnt(0);
#else
  asm volatile("s_wait_asynccnt 0" ::: "memory");
#endif
  __syncthreads();

  // ---- 4) K/V projections via WMMA bf16: 13 row-tiles over 8 waves ----
  for (int tile = wave; tile < NTILES; tile += 8) {
    const int row0 = tile * 16;
    // A fragment (16x32 bf16): lane<16 -> K chunks {0..7,16..23}; lane>=16 -> +8
    const int arow = row0 + (lane & 15);
    const int kb   = (lane < 16) ? 0 : 8;
    const float* ap = sAct + arow * COL;
    v16bf a0, a1;
    #pragma unroll
    for (int j = 0; j < 8; ++j) {
      a0[j]     = (__bf16)ap[kb + j];
      a0[j + 8] = (__bf16)ap[kb + 16 + j];
      a1[j]     = (__bf16)ap[32 + kb + j];
      a1[j + 8] = (__bf16)ap[32 + kb + 16 + j];
    }
    // C/D layout: lane = N (mod 16), VGPR r = M row (+8 for lanes 16..31)
    const int N  = lane & 15;
    const int mb = (lane < 16) ? 0 : 8;
    #pragma unroll
    for (int nt = 0; nt < 4; ++nt) {
      // Pull the 4 needed B fragments from LDS (2x ds_load_b128 each)
      const v16bf bk0 = *(const v16bf*)(sWK + ((0 * 4 + nt) * 512 + lane * 16));
      const v16bf bk1 = *(const v16bf*)(sWK + ((1 * 4 + nt) * 512 + lane * 16));
      const v16bf bv0 = *(const v16bf*)(sWV + ((0 * 4 + nt) * 512 + lane * 16));
      const v16bf bv1 = *(const v16bf*)(sWV + ((1 * 4 + nt) * 512 + lane * 16));
      v8f acck = {};
      v8f accv = {};
      // interleave K and V chains: independent op between dependent WMMAs
      acck = __builtin_amdgcn_wmma_f32_16x16x32_bf16(false, a0, false, bk0, (short)0, acck, false, false);
      accv = __builtin_amdgcn_wmma_f32_16x16x32_bf16(false, a0, false, bv0, (short)0, accv, false, false);
      acck = __builtin_amdgcn_wmma_f32_16x16x32_bf16(false, a1, false, bk1, (short)0, acck, false, false);
      accv = __builtin_amdgcn_wmma_f32_16x16x32_bf16(false, a1, false, bv1, (short)0, accv, false, false);
      #pragma unroll
      for (int r = 0; r < 8; ++r) {
        sKH[(row0 + mb + r) * COL + nt * 16 + N] = (__bf16)acck[r];
        sVH[(row0 + mb + r) * COL + nt * 16 + N] = (__bf16)accv[r];
      }
    }
  }
  __syncthreads();

  // ---- 5) scores = q . k^T / 8 ----
  for (int s = t; s < SK; s += 256) {
    const __bf16* kr = sKH + s * COL;
    #pragma unroll
    for (int h = 0; h < NH; ++h) {
      float dot = 0.f;
      #pragma unroll
      for (int d = 0; d < ATT; ++d) dot += sQ[h * ATT + d] * (float)kr[h * ATT + d];
      sSc[h * SK_PAD + s] = dot * 0.125f;
    }
  }
  __syncthreads();

  // ---- 6) per-head softmax over 200 keys (64 threads per head) ----
  const int h = t >> 6;
  const int g = t & 63;
  float m = -3.0e38f;
  for (int s = g; s < SK; s += 64) m = fmaxf(m, sSc[h * SK_PAD + s]);
  sRed[t] = m;
  __syncthreads();
  #pragma unroll
  for (int off = 32; off >= 1; off >>= 1) {
    if (g < off) sRed[t] = fmaxf(sRed[t], sRed[t + off]);
    __syncthreads();
  }
  const float hmax = sRed[h << 6];
  float ssum = 0.f;
  for (int s = g; s < SK; s += 64) {
    float e = __expf(sSc[h * SK_PAD + s] - hmax);
    sSc[h * SK_PAD + s] = e;
    ssum += e;
  }
  __syncthreads();
  sRed[t] = ssum;
  __syncthreads();
  #pragma unroll
  for (int off = 32; off >= 1; off >>= 1) {
    if (g < off) sRed[t] += sRed[t + off];
    __syncthreads();
  }
  const float inv = 1.0f / sRed[h << 6];
  for (int s = g; s < SK; s += 64) sSc[h * SK_PAD + s] *= inv;
  __syncthreads();

  // ---- 7) out = attn @ v  (64 cols x 4-way split reduction over s) ----
  const int o  = t & 63;
  const int ch = t >> 6;
  const int oh = o >> 4;
  float part = 0.f;
  for (int s = ch * 50; s < ch * 50 + 50; ++s)
    part += sSc[oh * SK_PAD + s] * (float)sVH[s * COL + o];
  sRed[t] = part;
  __syncthreads();
  if (t < 64)
    out[(size_t)b * 64 + t] = sRed[t] + sRed[t + 64] + sRed[t + 128] + sRed[t + 192];
}

extern "C" void kernel_launch(void* const* d_in, const int* in_sizes, int n_in,
                              void* d_out, int out_size, void* d_ws, size_t ws_size,
                              hipStream_t stream) {
  (void)n_in; (void)out_size; (void)d_ws; (void)ws_size;
  const float* qin = (const float*)d_in[0];
  const float* act = (const float*)d_in[1];
  const float* Qw  = (const float*)d_in[2];
  const float* Kw  = (const float*)d_in[3];
  const float* Vw  = (const float*)d_in[4];
  float* out = (float*)d_out;
  const int B = in_sizes[0] / COL;   // query_input is [B,1,64]
  (void)hipFuncSetAttribute(reinterpret_cast<const void*>(attn_fused),
                            hipFuncAttributeMaxDynamicSharedMemorySize, LDS_BYTES);
  attn_fused<<<B, 256, LDS_BYTES, stream>>>(qin, act, Qw, Kw, Vw, out);
}
